// FNet_22677427322923
// MI455X (gfx1250) — matrix operand order repaired and checked by reading, not compiled, not run on hardware
//
#include <hip/hip_runtime.h>

typedef _Float16 f16;
typedef __attribute__((ext_vector_type(16))) _Float16 v16h;
typedef __attribute__((ext_vector_type(8)))  _Float16 v8h;
typedef __attribute__((ext_vector_type(8)))  float    v8f;

#define DEV static __device__ __forceinline__

// ---------------------------------------------------------------------------
// WMMA fragment loaders (f16 16x16x32, wave32; A/B in this code's contiguous k order)
// A (16x32):  lane l -> row l&15 ; halves [0..15] = K 16*(l>>4)..+15
// B (32x16):  lane l -> col l&15 ; halves [0..15] = K 16*(l>>4)..+15
//             (contiguous in a row of W stored [out,in] row-major)
// C/D (16x16 f32): lane l, elem i -> row 8*(l>>4)+i, col l&15
// ---------------------------------------------------------------------------
DEV v16h load_fragA(const f16* base, int ld) {
  const int l = threadIdx.x & 31;
  const int row = l & 15, hi = l >> 4;
  const f16* p = base + (size_t)row * ld + hi * 16;
  union { v16h v; v8h h[2]; } u;
  u.h[0] = *(const v8h*)(p);
  u.h[1] = *(const v8h*)(p + 8);
  return u.v;
}

DEV v16h load_fragB(const f16* base, int ld) {
  const int l = threadIdx.x & 31;
  const int n = l & 15, hi = l >> 4;
  const f16* p = base + (size_t)n * ld + hi * 16;
  union { v16h v; v8h h[2]; } u;
  u.h[0] = *(const v8h*)(p);
  u.h[1] = *(const v8h*)(p + 8);
  return u.v;
}

DEV v8f wmma_f16f32(v16h a, v16h b, v8f c) {
  return __builtin_amdgcn_wmma_f32_16x16x32_f16(false, a, false, b, (short)0, c,
                                                false, false);
}

// Async global -> LDS DMA (CDNA5, tracked by ASYNCcnt). Per-lane 16B copy.
DEV void async_b128_to_lds(const void* lds_dst, const void* gsrc) {
  unsigned l = (unsigned)(unsigned long long)lds_dst;  // low 32b = LDS offset
  unsigned long long g = (unsigned long long)gsrc;
  asm volatile("global_load_async_to_lds_b128 %0, %1, off"
               :
               : "v"(l), "v"(g)
               : "memory");
}

// ---------------------------------------------------------------------------
// Generic GEMM:  C[M,N](f16) = act(A[M,K](f16) * W[N,K](f16)^T + bias(f32))
// Block: 256 threads = 8 waves stacked in M (block tile 128 x 128).
// Wave tile 16x128 (8 accumulators, 8 WMMA / 32-K step).
// W tile (128x32) double-buffered in LDS, filled with async global->LDS DMA
// issued one k-step ahead; s_wait_asynccnt + barrier consumes it.
// Requires: M % 128 == 0, N % 128 == 0, K % 32 == 0.
// ---------------------------------------------------------------------------
__global__ void __launch_bounds__(256) k_gemm(
    const f16* __restrict__ A, int lda,
    const f16* __restrict__ W, int ldw,
    const float* __restrict__ bias,
    f16* __restrict__ C, int ldc, int Ktot, int relu) {
  __shared__ __align__(16) f16 sW[2][128 * 32];
  const int wave = threadIdx.x >> 5;
  const int lane = threadIdx.x & 31;
  const int mBase = blockIdx.x * 128 + wave * 16;
  const int nBase = blockIdx.y * 128;

  // cooperative tile fill: 128 rows x 32 halves = 512 x 16B chunks,
  // 2 chunks per thread, one async DMA instruction per chunk
  auto issue_tile = [&](int k0, int buf) {
#pragma unroll
    for (int j = 0; j < 2; ++j) {
      const int c = threadIdx.x + j * 256;
      const int r = c >> 2, o = (c & 3) * 8;
      async_b128_to_lds(&sW[buf][r * 32 + o],
                        W + (size_t)(nBase + r) * ldw + k0 + o);
    }
  };

  v8f acc[8];
#pragma unroll
  for (int t = 0; t < 8; ++t) acc[t] = (v8f){};

  issue_tile(0, 0);
  const int nk = Ktot >> 5;
  for (int kk = 0; kk < nk; ++kk) {
    const int k0 = kk << 5;
    if (kk + 1 < nk) {
      issue_tile(k0 + 32, (kk + 1) & 1);
      asm volatile("s_wait_asynccnt 0x2" ::: "memory");  // tile kk landed
    } else {
      asm volatile("s_wait_asynccnt 0x0" ::: "memory");
    }
    __syncthreads();  // all waves' DMA portions visible
    v16h a = load_fragA(A + (size_t)mBase * lda + k0, lda);
    const f16* bufp = &sW[kk & 1][0];
#pragma unroll
    for (int t = 0; t < 8; ++t) {
      v16h b = load_fragB(bufp + t * 16 * 32, 32);
      acc[t] = wmma_f16f32(a, b, acc[t]);
    }
    __syncthreads();  // reads done before this buffer is DMA-overwritten
  }

  const int hi = lane >> 4, nn = lane & 15;
#pragma unroll
  for (int t = 0; t < 8; ++t) {
    const int col = nBase + t * 16 + nn;
    const float bv = bias ? bias[col] : 0.0f;
#pragma unroll
    for (int i = 0; i < 8; ++i) {
      float v = acc[t][i] + bv;
      if (relu) v = fmaxf(v, 0.0f);
      C[(size_t)(mBase + hi * 8 + i) * ldc + col] = (f16)v;
    }
  }
}

// ---------------------------------------------------------------------------
// Flash-attention: one wave = 16 query rows of one (b,h).
// Q,K: [B*2048, 512] f16 (head h = cols h*64..+63).  Vt: [B,H,64,2048] f16.
// O:   [B*2048, 512] f16.  grid = (2048/64, B*H), block = 128 (4 waves).
// ---------------------------------------------------------------------------
__global__ void __launch_bounds__(128) k_attn(
    const f16* __restrict__ Q, const f16* __restrict__ Km,
    const f16* __restrict__ Vt, f16* __restrict__ O) {
  __shared__ __align__(16) f16 sP[4][16 * 32];
  const int wave = threadIdx.x >> 5, lane = threadIdx.x & 31;
  const int hi = lane >> 4, nn = lane & 15;
  const int bh = blockIdx.y, b = bh >> 3, h = bh & 7;
  const int qBase = blockIdx.x * 64 + wave * 16;

  const f16* qp = Q + ((size_t)(b * 2048 + qBase)) * 512 + h * 64;
  const f16* kp = Km + ((size_t)(b * 2048)) * 512 + h * 64;
  const f16* vp = Vt + (size_t)bh * 64 * 2048;
  f16* sPw = &sP[wave][0];

  const v16h qa0 = load_fragA(qp, 512);        // K dims 0..31
  const v16h qa1 = load_fragA(qp + 32, 512);   // K dims 32..63

  v8f o[4];
#pragma unroll
  for (int t = 0; t < 4; ++t) o[t] = (v8f){};
  float mrow[8], lrow[8];
#pragma unroll
  for (int i = 0; i < 8; ++i) { mrow[i] = -1e30f; lrow[i] = 0.0f; }
  const float scale = 0.125f;  // 1/sqrt(64)

  for (int kb = 0; kb < 2048; kb += 32) {
    // hint-prefetch next key block toward the WGP (speculative, OOB-safe)
    __builtin_prefetch(kp + (size_t)(kb + 32 + nn) * 512 + hi * 32, 0, 1);
    v8f s0 = (v8f){}, s1 = (v8f){};
    {
      v16h b00 = load_fragB(kp + (size_t)kb * 512, 512);
      v16h b01 = load_fragB(kp + (size_t)kb * 512 + 32, 512);
      s0 = wmma_f16f32(qa0, b00, s0);
      s0 = wmma_f16f32(qa1, b01, s0);
      v16h b10 = load_fragB(kp + (size_t)(kb + 16) * 512, 512);
      v16h b11 = load_fragB(kp + (size_t)(kb + 16) * 512 + 32, 512);
      s1 = wmma_f16f32(qa0, b10, s1);
      s1 = wmma_f16f32(qa1, b11, s1);
    }
    float co[8];
#pragma unroll
    for (int i = 0; i < 8; ++i) {
      float a0 = s0[i] * scale, a1 = s1[i] * scale;
      // row max across the 16 lanes of this half-wave (xor bits 0..3 only)
      float v = fmaxf(a0, a1);
      v = fmaxf(v, __shfl_xor(v, 1));
      v = fmaxf(v, __shfl_xor(v, 2));
      v = fmaxf(v, __shfl_xor(v, 4));
      v = fmaxf(v, __shfl_xor(v, 8));
      const float mn = fmaxf(mrow[i], v);
      const float p0 = __expf(a0 - mn), p1 = __expf(a1 - mn);
      const int r = hi * 8 + i;
      sPw[r * 32 + nn] = (f16)p0;        // keys kb + nn
      sPw[r * 32 + nn + 16] = (f16)p1;   // keys kb + 16 + nn
      float rs = p0 + p1;
      rs += __shfl_xor(rs, 1);
      rs += __shfl_xor(rs, 2);
      rs += __shfl_xor(rs, 4);
      rs += __shfl_xor(rs, 8);
      const float corr = __expf(mrow[i] - mn);
      lrow[i] = lrow[i] * corr + rs;
      mrow[i] = mn;
      co[i] = corr;
    }
#pragma unroll
    for (int t = 0; t < 4; ++t)
#pragma unroll
      for (int i = 0; i < 8; ++i) o[t][i] *= co[i];
    // P (16x32, D-layout in LDS) re-read as an A-fragment; same-wave LDS
    // ops are in-order so no barrier is required.
    v16h pa = load_fragA(sPw, 32);
#pragma unroll
    for (int t = 0; t < 4; ++t) {
      v16h bv = load_fragB(vp + (size_t)(t * 16) * 2048 + kb, 2048);
      o[t] = wmma_f16f32(pa, bv, o[t]);
    }
  }

  f16* op = O + ((size_t)(b * 2048 + qBase)) * 512 + h * 64;
#pragma unroll
  for (int t = 0; t < 4; ++t)
#pragma unroll
    for (int i = 0; i < 8; ++i) {
      float v = o[t][i] / lrow[i];
      op[(size_t)(hi * 8 + i) * 512 + t * 16 + nn] = (f16)v;
    }
}

// ---------------------------------------------------------------------------
// Small helper kernels
// ---------------------------------------------------------------------------
__global__ void k_f32_to_f16(const float* __restrict__ s, f16* __restrict__ d,
                             int n) {
  int i = blockIdx.x * blockDim.x + threadIdx.x;
  if (i < n) d[i] = (f16)s[i];
}

__global__ void k_pad_rows(const float* __restrict__ s, int sr, int sc,
                           f16* __restrict__ d, int dr, int dc) {
  int i = blockIdx.x * blockDim.x + threadIdx.x;
  if (i >= dr * dc) return;
  int r = i / dc, c = i % dc;
  float v = (r < sr && c < sc) ? s[r * sc + c] : 0.0f;
  d[i] = (f16)v;
}

__global__ void k_pad_bias(const float* __restrict__ s, float* __restrict__ d,
                           int sn, int dn) {
  int i = blockIdx.x * blockDim.x + threadIdx.x;
  if (i < dn) d[i] = (i < sn) ? s[i] : 0.0f;
}

// points: flat f32 [4096*32, 3] per cloud; chunk of 16384 points -> [16384,32]
__global__ void k_pad_points(const float* __restrict__ pts, int pointBase,
                             f16* __restrict__ out) {
  int p = blockIdx.x * blockDim.x + threadIdx.x;  // 0..16383
  const float* s = pts + (size_t)(pointBase + p) * 3;
  __align__(16) f16 buf[32];
  buf[0] = (f16)s[0];
  buf[1] = (f16)s[1];
  buf[2] = (f16)s[2];
#pragma unroll
  for (int i = 3; i < 32; ++i) buf[i] = (f16)0.0f;
  v8h* d = (v8h*)(out + (size_t)p * 32);
#pragma unroll
  for (int i = 0; i < 4; ++i) d[i] = *(v8h*)(buf + i * 8);
}

// in: [512 groups * 32 pts, 512] f16 -> max over the 32 points -> out[512,512]
__global__ void k_maxk(const f16* __restrict__ in, f16* __restrict__ out) {
  int idx = blockIdx.x * blockDim.x + threadIdx.x;  // 512*512
  int g = idx >> 9, d = idx & 511;
  float m = -1e30f;
  const f16* p = in + ((size_t)g * 32) * 512 + d;
#pragma unroll 4
  for (int k = 0; k < 32; ++k) m = fmaxf(m, (float)p[(size_t)k * 512]);
  out[idx] = (f16)m;
}

// V [B*2048, 512] (b,n,h,dh) -> Vt [B,H,64,2048]
__global__ void k_transpose_v(const f16* __restrict__ V, f16* __restrict__ Vt) {
  int i = blockIdx.x * blockDim.x + threadIdx.x;  // 2*8*64*2048
  int n = i & 2047, d = (i >> 11) & 63, h = (i >> 17) & 7, b = i >> 20;
  Vt[i] = V[((size_t)(b * 2048 + n)) * 512 + h * 64 + d];
}

// head l3: [4096, 128(pad, use 32)] f16 x w3[2,32] f32 + b3 -> f32 [4096,2]
__global__ void k_head3(const f16* __restrict__ hbuf,
                        const float* __restrict__ w,
                        const float* __restrict__ b, float* __restrict__ out) {
  int r = blockIdx.x * blockDim.x + threadIdx.x;  // 4096
  const f16* x = hbuf + (size_t)r * 128;
  float a0 = b[0], a1 = b[1];
#pragma unroll
  for (int c = 0; c < 32; ++c) {
    float xv = (float)x[c];
    a0 += xv * w[c];
    a1 += xv * w[32 + c];
  }
  out[r * 2 + 0] = a0;
  out[r * 2 + 1] = a1;
}

// ---------------------------------------------------------------------------
// Host orchestration
// ---------------------------------------------------------------------------
extern "C" void kernel_launch(void* const* d_in, const int* in_sizes, int n_in,
                              void* d_out, int out_size, void* d_ws,
                              size_t ws_size, hipStream_t stream) {
  (void)in_sizes; (void)n_in; (void)out_size; (void)ws_size;

  // ---- input indices (setup_inputs dict order, depth-first) ----
  const float* src = (const float*)d_in[0];
  const float* tgt = (const float*)d_in[1];
  const float* p1 = (const float*)d_in[2];
  const float* hw2 = (const float*)d_in[24];
  const float* hb2 = (const float*)d_in[25];
  const float* hw3 = (const float*)d_in[26];
  const float* hb3 = (const float*)d_in[27];

  // ---- workspace carve-up ----
  size_t off = 0;
  auto alloc = [&](size_t bytes) -> void* {
    void* p = (char*)d_ws + off;
    off = (off + bytes + 255) & ~(size_t)255;
    return p;
  };
  f16* w1h = (f16*)alloc(256 * 32 * 2);
  f16* w2h = (f16*)alloc(256 * 256 * 2);
  f16* w3h = (f16*)alloc(512 * 256 * 2);
  f16* w4h = (f16*)alloc(512 * 512 * 2);
  f16* attW[8];  // enc wq,wk,wv,wo, dec wq,wk,wv,wo
  for (int i = 0; i < 8; ++i) attW[i] = (f16*)alloc(512 * 512 * 2);
  f16* hw1h = (f16*)alloc(128 * 512 * 2);
  f16* hw2h = (f16*)alloc(128 * 128 * 2);
  float* hb2p = (float*)alloc(128 * 4);
  f16* ptpad = (f16*)alloc((size_t)16384 * 32 * 2);
  f16* l1o = (f16*)alloc((size_t)16384 * 256 * 2);
  f16* l2o = (f16*)alloc((size_t)16384 * 256 * 2);
  f16* l3o = (f16*)alloc((size_t)16384 * 512 * 2);
  f16* l4o = (f16*)alloc((size_t)16384 * 512 * 2);
  f16* xf = (f16*)alloc((size_t)4096 * 512 * 2);
  f16* yf = (f16*)alloc((size_t)4096 * 512 * 2);
  f16* qb = (f16*)alloc((size_t)4096 * 512 * 2);
  f16* kb = (f16*)alloc((size_t)4096 * 512 * 2);
  f16* vb = (f16*)alloc((size_t)4096 * 512 * 2);
  f16* vtb = (f16*)alloc((size_t)4096 * 512 * 2);
  f16* ob = (f16*)alloc((size_t)4096 * 512 * 2);
  f16* x2 = (f16*)alloc((size_t)4096 * 512 * 2);
  f16* y2 = (f16*)alloc((size_t)4096 * 512 * 2);
  f16* xm = (f16*)alloc((size_t)4096 * 512 * 2);
  f16* ym = (f16*)alloc((size_t)4096 * 512 * 2);
  f16* h1o = (f16*)alloc((size_t)4096 * 128 * 2);
  f16* h2o = (f16*)alloc((size_t)4096 * 128 * 2);

  auto cdiv = [](int a, int b) { return (a + b - 1) / b; };
  auto conv = [&](int idx, f16* dst, int n) {
    k_f32_to_f16<<<cdiv(n, 256), 256, 0, stream>>>((const float*)d_in[idx],
                                                   dst, n);
  };
  auto gemm = [&](const f16* A, int lda, const f16* W, int ldw,
                  const float* bias, f16* C, int ldc, int M, int N, int K,
                  int relu) {
    dim3 g(M / 128, N / 128);
    k_gemm<<<g, 256, 0, stream>>>(A, lda, W, ldw, bias, C, ldc, K, relu);
  };

  // ---- weight conversion / padding ----
  conv(3, w2h, 256 * 256);
  conv(4, w3h, 512 * 256);
  conv(5, w4h, 512 * 512);
  const int attIdx[8] = {6, 8, 10, 12, 14, 16, 18, 20};
  for (int i = 0; i < 8; ++i) conv(attIdx[i], attW[i], 512 * 512);
  conv(22, hw1h, 128 * 512);
  k_pad_rows<<<cdiv(256 * 32, 256), 256, 0, stream>>>(p1, 256, 3, w1h, 256, 32);
  k_pad_rows<<<cdiv(128 * 128, 256), 256, 0, stream>>>(hw2, 32, 128, hw2h, 128,
                                                       128);
  k_pad_bias<<<1, 128, 0, stream>>>(hb2, hb2p, 32, 128);

  // ---- PointNet (chunked: 8 chunks of 512 groups = 16384 points/chunk) ----
  for (int cloud = 0; cloud < 2; ++cloud) {
    const float* pts = cloud ? tgt : src;
    f16* feat = cloud ? yf : xf;
    for (int c = 0; c < 8; ++c) {
      const int gBase = c * 512;
      k_pad_points<<<16384 / 256, 256, 0, stream>>>(pts, gBase * 32, ptpad);
      gemm(ptpad, 32, w1h, 32, nullptr, l1o, 256, 16384, 256, 32, 1);
      gemm(l1o, 256, w2h, 256, nullptr, l2o, 256, 16384, 256, 256, 1);
      gemm(l2o, 256, w3h, 256, nullptr, l3o, 512, 16384, 512, 256, 1);
      gemm(l3o, 512, w4h, 512, nullptr, l4o, 512, 16384, 512, 512, 1);
      k_maxk<<<(512 * 512) / 256, 256, 0, stream>>>(l4o,
                                                    feat + (size_t)gBase * 512);
    }
  }

  // ---- MHA (4 calls: enc self x, enc self y, dec cross, dec cross) ----
  auto run_mha = [&](const f16* qs, const f16* kvs, f16* const* w4p,
                     const float* bq, const float* bk2, const float* bv2,
                     const float* bo, f16* out) {
    gemm(qs, 512, w4p[0], 512, bq, qb, 512, 4096, 512, 512, 0);
    gemm(kvs, 512, w4p[1], 512, bk2, kb, 512, 4096, 512, 512, 0);
    gemm(kvs, 512, w4p[2], 512, bv2, vb, 512, 4096, 512, 512, 0);
    k_transpose_v<<<(2 * 8 * 64 * 2048) / 256, 256, 0, stream>>>(vb, vtb);
    k_attn<<<dim3(32, 16), 128, 0, stream>>>(qb, kb, vtb, ob);
    gemm(ob, 512, w4p[3], 512, bo, out, 512, 4096, 512, 512, 0);
  };
  const float* eb[4] = {(const float*)d_in[7], (const float*)d_in[9],
                        (const float*)d_in[11], (const float*)d_in[13]};
  const float* db[4] = {(const float*)d_in[15], (const float*)d_in[17],
                        (const float*)d_in[19], (const float*)d_in[21]};
  run_mha(xf, xf, &attW[0], eb[0], eb[1], eb[2], eb[3], x2);
  run_mha(yf, yf, &attW[0], eb[0], eb[1], eb[2], eb[3], y2);
  run_mha(x2, y2, &attW[4], db[0], db[1], db[2], db[3], xm);
  run_mha(y2, x2, &attW[4], db[0], db[1], db[2], db[3], ym);

  // ---- head (xm -> out[0:8192], ym -> out[8192:16384]) ----
  auto run_head = [&](const f16* in, float* out) {
    gemm(in, 512, hw1h, 512, (const float*)d_in[23], h1o, 128, 4096, 128, 512,
         1);
    gemm(h1o, 128, hw2h, 128, hb2p, h2o, 128, 4096, 128, 128, 1);
    k_head3<<<4096 / 256, 256, 0, stream>>>(h2o, hw3, hb3, out);
  };
  run_head(xm, (float*)d_out);
  run_head(ym, (float*)d_out + 8192);
}
